// VIN_38156489458223
// MI455X (gfx1250) — matrix operand order repaired and checked by reading, not compiled, not run on hardware
//
#include <hip/hip_runtime.h>

typedef __attribute__((ext_vector_type(16))) _Float16 v16h;
typedef __attribute__((ext_vector_type(8)))  _Float16 v8h;
typedef __attribute__((ext_vector_type(4)))  _Float16 v4h;
typedef __attribute__((ext_vector_type(8)))  float    v8f;

#define IMG   8
#define LH    150
#define LQ    10
#define KITER 10
#define NPIX  64
#define PADW  16      // padded row stride in halves (32 B, b128-friendly)
#define C0    4       // interior starts at padded column 4
#define PIMG  (10*PADW)   // 160 halves per padded image
#define IMGB  (16*PIMG*2) // 5120 B: one f16 image set for 16 batches
#define BPW   16
#define WAVES 4
#define BPB   (BPW*WAVES)
#define BTOT  16384

// patch element e (0..15) of a 16-half row held as two v8h
#define EL(A, B, E) (((E) < 8) ? (A)[(E)] : (B)[(E) - 8])

__device__ __forceinline__ float swap16_max(float m) {
  int mi = __builtin_bit_cast(int, m);
#if __has_builtin(__builtin_amdgcn_permlanex16)
  int sw = __builtin_amdgcn_permlanex16(mi, mi, 0x76543210, 0xFEDCBA98, false, false);
#else
  int sw = __builtin_amdgcn_ds_swizzle(mi, 0x401F);  // SWAPX16
#endif
  return fmaxf(m, __builtin_bit_cast(float, sw));
}

// ws layout: float[0..8]=Weff, float[9]=beff; byte 64: Wcat A-fragment (32 lanes x 16 halves)

__global__ void vin_prep(const float* __restrict__ Wh, const float* __restrict__ bh,
                         const float* __restrict__ Wr, const float* __restrict__ Wq,
                         const float* __restrict__ w,  float* __restrict__ ws) {
  int tid = threadIdx.x;
  if (tid < 10) {
    float acc = 0.f;
    if (tid < 9) { for (int c = 0; c < LH; ++c) acc += Wr[c] * Wh[c * 9 + tid]; }
    else         { for (int c = 0; c < LH; ++c) acc += Wr[c] * bh[c]; }
    ws[tid] = acc;
  }
  if (tid < 32) {
    _Float16* af = (_Float16*)((char*)ws + 64) + tid * 16;
    int  ch = tid & 15;
    bool hi = tid >= 16;
    for (int i = 0; i < 16; ++i) af[i] = (_Float16)0.f;
    if (ch < LQ) {
      if (!hi) {   // lanes 0-15: elements 0-8 = Wq taps 0-8 (pair with r)
        for (int t = 0; t < 9; ++t) af[t] = (_Float16)Wq[ch * 9 + t];
      } else {     // lanes 16-31: elements 0-8 = w taps 0-8 (pair with v)
        for (int t = 0; t < 9; ++t) af[t] = (_Float16)w [ch * 9 + t];
      }
    }
  }
}

__global__ void __launch_bounds__(WAVES * 32)
vin_main(const float* __restrict__ S, const float* __restrict__ Wfc,
         const float* __restrict__ ws, float* __restrict__ out) {
  extern __shared__ char smem[];
  const int tid  = threadIdx.x;
  const int lane = tid & 31;
  const int wave = tid >> 5;
  const int b0   = blockIdx.x * BPB + wave * BPW;

  _Float16* wbase = (_Float16*)(smem + wave * (3 * IMGB));
  _Float16* Rp = wbase;
  _Float16* V0 = wbase + 1 * BPW * PIMG;
  _Float16* V1 = wbase + 2 * BPW * PIMG;
  float* WfcS  = (float*)(smem + WAVES * (3 * IMGB));
  float* qsel  = WfcS + 80 + wave * (BPW * LQ);

  // zero the wave's 3 images (15360 B = 960 int4)
  {
    int4 z = {0, 0, 0, 0};
    int4* p = (int4*)wbase;
    for (int i = lane; i < 960; i += 32) p[i] = z;
  }
  if (tid < 80) WfcS[tid] = Wfc[tid];

  // stage X (f16) into V0 interior
  for (int i = lane; i < BPW * NPIX; i += 32) {
    int b = i >> 6, p = i & 63, y = p >> 3, x = p & 7;
    V0[b * PIMG + (y + 1) * PADW + (C0 + x)] = (_Float16)S[(size_t)(b0 + b) * 66 + p];
  }

  float weff[9];
#pragma unroll
  for (int t = 0; t < 9; ++t) weff[t] = ws[t];
  float beff = ws[9];

  v16h afrag = *(const v16h*)((const char*)ws + 64 + lane * 32);

  const int bl = lane & 15;
  const int s1 = (int)S[(size_t)(b0 + bl) * 66 + 64];
  const int s2 = (int)S[(size_t)(b0 + bl) * 66 + 65];
  const int tpix = s1 * 8 + s2;
  const bool hiHalf = (lane >= 16);

  __syncthreads();

  // r = conv(X, Weff, pad=1) + beff   (fused Wh/Wr/bh hidden layer)
  for (int i = lane; i < BPW * NPIX; i += 32) {
    int b = i >> 6, p = i & 63, y = p >> 3, x = p & 7;
    const _Float16* src = V0 + b * PIMG + y * PADW + (C0 + x - 1);
    float acc = beff;
    acc += weff[0] * (float)src[0]  + weff[1] * (float)src[1]  + weff[2] * (float)src[2];
    acc += weff[3] * (float)src[16] + weff[4] * (float)src[17] + weff[5] * (float)src[18];
    acc += weff[6] * (float)src[32] + weff[7] * (float)src[33] + weff[8] * (float)src[34];
    Rp[b * PIMG + (y + 1) * PADW + (C0 + x)] = (_Float16)acc;
  }
  __syncthreads();

  // re-zero V0: initial v == 0
  {
    int4 z = {0, 0, 0, 0};
    int4* p = (int4*)V0;
    for (int i = lane; i < 320; i += 32) p[i] = z;
  }
  __syncthreads();

  _Float16* vcur = V0;
  _Float16* vnxt = V1;

  // ---- hot loop: 10 value-iteration passes, no selection logic ----
  for (int pass = 0; pass < KITER; ++pass) {
    const _Float16* base = (hiHalf ? vcur : Rp) + bl * PIMG;
    // rolling row registers: only 2 new ds_load_b128 per output row
    v8h a0 = *(const v8h*)(base);
    v8h c0v = *(const v8h*)(base + 8);
    v8h a1 = *(const v8h*)(base + PADW);
    v8h c1v = *(const v8h*)(base + PADW + 8);
    for (int y = 0; y < 8; ++y) {
      v8h a2 = *(const v8h*)(base + (y + 2) * PADW);
      v8h c2v = *(const v8h*)(base + (y + 2) * PADW + 8);
      v4h nvlo = {}, nvhi = {};
#pragma unroll
      for (int x = 0; x < 8; ++x) {
        v16h bf = {};
        bf[0] = EL(a0, c0v, C0 + x - 1);
        bf[1] = EL(a0, c0v, C0 + x);
        bf[2] = EL(a0, c0v, C0 + x + 1);
        bf[3] = EL(a1, c1v, C0 + x - 1);
        bf[4] = EL(a1, c1v, C0 + x);
        bf[5] = EL(a1, c1v, C0 + x + 1);
        bf[6] = EL(a2, c2v, C0 + x - 1);
        bf[7] = EL(a2, c2v, C0 + x);
        bf[8] = EL(a2, c2v, C0 + x + 1);
        v8f c = {};
        v8f d = __builtin_amdgcn_wmma_f32_16x16x32_f16(false, afrag, false, bf,
                                                       (short)0, c, false, false);
        float m01  = fmaxf(d[0], d[1]);
        float full = fmaxf(fmaxf(m01, fmaxf(d[2], d[3])),
                           fmaxf(fmaxf(d[4], d[5]), fmaxf(d[6], d[7])));
        float m = swap16_max(hiHalf ? m01 : full);
        if (x < 4) nvlo[x] = (_Float16)m; else nvhi[x - 4] = (_Float16)m;
      }
      if (!hiHalf) {
        _Float16* dst = vnxt + bl * PIMG + (y + 1) * PADW + C0;  // 8B-aligned
        *(v4h*)dst       = nvlo;
        *(v4h*)(dst + 4) = nvhi;
      }
      a0 = a1; c0v = c1v; a1 = a2; c1v = c2v;
    }
    _Float16* t = vcur; vcur = vnxt; vnxt = t;
    __syncthreads();
  }

  // ---- final pass: only the pixels some batch element actually selects ----
  {
    const _Float16* base = (hiHalf ? vcur : Rp) + bl * PIMG;
    for (int p = 0; p < NPIX; ++p) {
      if (__ballot(tpix == p) == 0ull) continue;   // wave-uniform skip
      int y = p >> 3, x = p & 7;
      const _Float16* s = base + y * PADW + (C0 + x - 1);
      v16h bf = {};
      bf[0] = s[0];  bf[1] = s[1];  bf[2] = s[2];
      bf[3] = s[16]; bf[4] = s[17]; bf[5] = s[18];
      bf[6] = s[32]; bf[7] = s[33]; bf[8] = s[34];
      v8f c = {};
      v8f d = __builtin_amdgcn_wmma_f32_16x16x32_f16(false, afrag, false, bf,
                                                     (short)0, c, false, false);
      if (tpix == p) {
        if (!hiHalf) {
#pragma unroll
          for (int g = 0; g < 8; ++g) qsel[bl * LQ + g] = d[g];
        } else {
          qsel[bl * LQ + 8] = d[0];
          qsel[bl * LQ + 9] = d[1];
        }
      }
    }
  }
  __syncthreads();

  // logits = q_sel @ Wfc^T (10 -> 8)
  if (lane < 16) {
    float q[10];
#pragma unroll
    for (int cc = 0; cc < 10; ++cc) q[cc] = qsel[lane * LQ + cc];
#pragma unroll
    for (int o = 0; o < 8; ++o) {
      float acc = 0.f;
#pragma unroll
      for (int cc = 0; cc < 10; ++cc) acc += q[cc] * WfcS[o * 10 + cc];
      out[(size_t)(b0 + lane) * 8 + o] = acc;
    }
  }
}

extern "C" void kernel_launch(void* const* d_in, const int* in_sizes, int n_in,
                              void* d_out, int out_size, void* d_ws, size_t ws_size,
                              hipStream_t stream) {
  const float* S   = (const float*)d_in[0];
  const float* Wh  = (const float*)d_in[1];
  const float* bh  = (const float*)d_in[2];
  const float* Wr  = (const float*)d_in[3];
  const float* Wq  = (const float*)d_in[4];
  const float* w   = (const float*)d_in[5];
  const float* Wfc = (const float*)d_in[6];
  float* ws   = (float*)d_ws;
  float* outp = (float*)d_out;

  vin_prep<<<1, 64, 0, stream>>>(Wh, bh, Wr, Wq, w, ws);

  const int blocks = BTOT / BPB;  // 256
  const size_t smemBytes = WAVES * (3 * IMGB)
                         + 80 * sizeof(float)
                         + WAVES * BPW * LQ * sizeof(float);
  vin_main<<<blocks, WAVES * 32, smemBytes, stream>>>(S, Wfc, ws, outp);
}
